// Int4Linear_52072183497436
// MI455X (gfx1250) — matrix operand order repaired and checked by reading, not compiled, not run on hardware
//
#include <hip/hip_runtime.h>

// ---- CDNA5 WMMA vector types (wave32) ----
typedef __attribute__((ext_vector_type(16))) _Float16 v16h;
typedef __attribute__((ext_vector_type(8)))  _Float16 v8h;
typedef __attribute__((ext_vector_type(8)))  float    v8f;
typedef __attribute__((ext_vector_type(4)))  unsigned u32x4;
typedef __attribute__((ext_vector_type(8)))  unsigned u32x8;

#define BM 128
#define BN 128
#define BK 64
#define K_DIM 4096
#define N_DIM 11008
#define PK_DIM (K_DIM / 2)   // packed int32 per output row
#define NG_DIM (K_DIM / 128) // quant groups per row
#define TILE_BYTES (BM * BK * 2)   // 16 KB per tile buffer

static __device__ __forceinline__ unsigned pack2_f16(float a, float b) {
  _Float16 ha = (_Float16)a;
  _Float16 hb = (_Float16)b;
  unsigned short ua = __builtin_bit_cast(unsigned short, ha);
  unsigned short ub = __builtin_bit_cast(unsigned short, hb);
  return (unsigned)ua | ((unsigned)ub << 16);
}

static __device__ __forceinline__ unsigned lds_off32(const void* p) {
  // ISA 10.2: flat->LDS mapping truncates to addr[31:0] (LDS byte offset).
  return (unsigned)(unsigned long long)(size_t)p;
}

// ---- CDNA5 async copy global->LDS (ASYNCcnt), GVS addressing ----
static __device__ __forceinline__ void async_load_b128(unsigned lds_byte_off,
                                                       unsigned voff,
                                                       const void* sbase) {
  asm volatile("global_load_async_to_lds_b128 %0, %1, %2"
               :: "v"(lds_byte_off), "v"(voff), "s"(sbase)
               : "memory");
}

static __device__ __forceinline__ void wait_asynccnt0() {
  asm volatile("s_wait_asynccnt 0x0" ::: "memory");
}

// ---- CDNA5 Tensor Data Mover: 2D tile load, D# groups 0+1 (2D form) ----
// Moves tile_d1 rows of tile_d0 elements (2B each) from a row-major tensor
// with row stride stride0 (elements) into LDS (rows packed at tile_d0*2 B).
static __device__ __forceinline__ void tdm_load_2d(unsigned lds_off,
                                                   unsigned long long gaddr,
                                                   unsigned tensor_d0,
                                                   unsigned tensor_d1,
                                                   unsigned tile_d0,
                                                   unsigned tile_d1,
                                                   unsigned long long stride0) {
  u32x4 g0;
  g0[0] = 1u;                                    // count=1 valid user D#
  g0[1] = lds_off;                               // lds_addr (bytes)
  g0[2] = (unsigned)gaddr;                       // global_addr[31:0]
  g0[3] = ((unsigned)(gaddr >> 32) & 0x01FFFFFFu) | (2u << 30); // [56:32]|type=2
  u32x8 g1;
  g1[0] = 1u << 16;                              // wg_mask=0, data_size=2B
  g1[1] = (tensor_d0 & 0xFFFFu) << 16;           // tensor_dim0[15:0] @bit48
  g1[2] = (tensor_d0 >> 16) | ((tensor_d1 & 0xFFFFu) << 16);
  g1[3] = (tensor_d1 >> 16) | ((tile_d0 & 0xFFFFu) << 16);
  g1[4] = tile_d1 & 0xFFFFu;                     // tile_dim1; tile_dim2=0
  g1[5] = (unsigned)stride0;                     // dim0 stride (elements)
  g1[6] = (unsigned)(stride0 >> 32) & 0xFFFFu;   // stride hi; dim1_stride=0
  g1[7] = 0u;
  asm volatile("tensor_load_to_lds %0, %1" :: "s"(g0), "s"(g1) : "memory");
}

// ============================================================
// Pass 1a: dequant int4 W -> f16  [N, K] row-major
// ============================================================
__global__ __launch_bounds__(256)
void dequant_w_kernel(const int* __restrict__ wp,
                      const float* __restrict__ scale,
                      const float* __restrict__ zp,
                      _Float16* __restrict__ w16) {
  const int o   = blockIdx.x;
  const int tid = threadIdx.x;
  const int j0  = tid * 8;              // first packed int32
  const int g   = j0 >> 6;              // 64 int32 per 128-element group
  const float s = scale[(size_t)o * NG_DIM + g];
  const float z = zp[(size_t)o * NG_DIM + g];

  const int* rp = wp + (size_t)o * PK_DIM + j0;
  const int4 p0 = *(const int4*)(rp);
  const int4 p1 = *(const int4*)(rp + 4);
  const int pv[8] = {p0.x, p0.y, p0.z, p0.w, p1.x, p1.y, p1.z, p1.w};

  unsigned oo[8];
#pragma unroll
  for (int e = 0; e < 8; ++e) {
    // unpack_order 'high_low': high nibble = even k, low nibble = odd k
    const float hi = (float)((pv[e] >> 4) & 15);
    const float lo = (float)(pv[e] & 15);
    oo[e] = pack2_f16((hi - z) * s, (lo - z) * s);
  }
  _Float16* orow = w16 + (size_t)o * K_DIM + (size_t)j0 * 2;
  *(uint4*)(orow)     = make_uint4(oo[0], oo[1], oo[2], oo[3]);
  *(uint4*)(orow + 8) = make_uint4(oo[4], oo[5], oo[6], oo[7]);
}

// ============================================================
// Pass 1b: x f32 -> f16, elementwise, 8 per thread
// ============================================================
__global__ __launch_bounds__(256)
void cvt_x_kernel(const float* __restrict__ x, _Float16* __restrict__ xh) {
  const size_t i = ((size_t)blockIdx.x * 256 + threadIdx.x) * 8;
  const float4 a = *(const float4*)(x + i);
  const float4 b = *(const float4*)(x + i + 4);
  *(uint4*)(xh + i) = make_uint4(pack2_f16(a.x, a.y), pack2_f16(a.z, a.w),
                                 pack2_f16(b.x, b.y), pack2_f16(b.z, b.w));
}

// ============================================================
// Shared compute helper: one K-tile of WMMAs from an LDS buffer.
// All 6 fragments of a K-step are loaded up front (distinct live values)
// so the scheduler can batch the 12 ds_load_b128 and use partial
// s_wait_dscnt thresholds instead of full drains.
// ============================================================
static __device__ __forceinline__ v16h load_afrag(const _Float16* ap) {
  const v8h alo = *(const v8h*)ap;
  const v8h ahi = *(const v8h*)(ap + 8);
  return __builtin_shufflevector(alo, ahi, 0, 1, 2, 3, 4, 5, 6, 7, 8, 9, 10,
                                 11, 12, 13, 14, 15);
}

static __device__ __forceinline__ void compute_tile(const _Float16* xl,
                                                    const _Float16* wl,
                                                    int wm, int wn, int half,
                                                    int r, v8f acc[4][2]) {
#pragma unroll
  for (int kk = 0; kk < BK; kk += 32) {
    const _Float16* bbase = wl + (wn * 32 + r) * BK + kk + half * 16;
    const _Float16* abase = xl + (wm * 64 + r) * BK + kk + half * 16;
    const v16h b0 = *(const v16h*)(bbase);
    const v16h b1 = *(const v16h*)(bbase + 16 * BK);
    const v16h a0 = load_afrag(abase);
    const v16h a1 = load_afrag(abase + 16 * BK);
    const v16h a2 = load_afrag(abase + 32 * BK);
    const v16h a3 = load_afrag(abase + 48 * BK);
    acc[0][0] = __builtin_amdgcn_wmma_f32_16x16x32_f16(
        false, a0, false, b0, (short)0, acc[0][0], false, false);
    acc[0][1] = __builtin_amdgcn_wmma_f32_16x16x32_f16(
        false, a0, false, b1, (short)0, acc[0][1], false, false);
    acc[1][0] = __builtin_amdgcn_wmma_f32_16x16x32_f16(
        false, a1, false, b0, (short)0, acc[1][0], false, false);
    acc[1][1] = __builtin_amdgcn_wmma_f32_16x16x32_f16(
        false, a1, false, b1, (short)0, acc[1][1], false, false);
    acc[2][0] = __builtin_amdgcn_wmma_f32_16x16x32_f16(
        false, a2, false, b0, (short)0, acc[2][0], false, false);
    acc[2][1] = __builtin_amdgcn_wmma_f32_16x16x32_f16(
        false, a2, false, b1, (short)0, acc[2][1], false, false);
    acc[3][0] = __builtin_amdgcn_wmma_f32_16x16x32_f16(
        false, a3, false, b0, (short)0, acc[3][0], false, false);
    acc[3][1] = __builtin_amdgcn_wmma_f32_16x16x32_f16(
        false, a3, false, b1, (short)0, acc[3][1], false, false);
  }
}

static __device__ __forceinline__ void store_tile(float* __restrict__ out,
                                                  const float* __restrict__ bias,
                                                  int m0, int n0, int wm, int wn,
                                                  int half, int r,
                                                  const v8f acc[4][2]) {
#pragma unroll
  for (int tn = 0; tn < 2; ++tn) {
    const int cn = n0 + wn * 32 + tn * 16 + r;
    const float bv = bias[cn];
#pragma unroll
    for (int tm = 0; tm < 4; ++tm) {
      const int rm = m0 + wm * 64 + tm * 16 + half * 8;
#pragma unroll
      for (int v = 0; v < 8; ++v) {
        out[(size_t)(rm + v) * N_DIM + cn] = acc[tm][tn][v] + bv;
      }
    }
  }
}

// ============================================================
// Pass 2 (primary): f16 GEMM, TDM-staged, 4 LDS buffers, depth-3 pipeline
// Wave 0 drives the Tensor Data Mover; all 8 waves compute.
// ============================================================
__global__ __launch_bounds__(256, 2)
void gemm_f16_wmma_tdm_kernel(const _Float16* __restrict__ xh,
                              const _Float16* __restrict__ wh,
                              const float* __restrict__ bias,
                              float* __restrict__ out, int M) {
  __shared__ __align__(32) _Float16 xsm[4][BM * BK];  // 4 x 16 KB
  __shared__ __align__(32) _Float16 wsm[4][BN * BK];  // 4 x 16 KB

  const int tid  = threadIdx.x;
  const int m0   = blockIdx.y * BM;
  const int n0   = blockIdx.x * BN;
  const int lane = tid & 31;
  const int wid  = tid >> 5;
  const int wm   = wid >> 2;
  const int wn   = wid & 3;
  const int half = lane >> 4;
  const int r    = lane & 15;

  v8f acc[4][2];
  const v8f vzero = {0.f, 0.f, 0.f, 0.f, 0.f, 0.f, 0.f, 0.f};
#pragma unroll
  for (int i = 0; i < 4; ++i)
#pragma unroll
    for (int j = 0; j < 2; ++j)
      acc[i][j] = vzero;

  const unsigned xlds = lds_off32(&xsm[0][0]);
  const unsigned wlds = lds_off32(&wsm[0][0]);
  const unsigned long long xga =
      (unsigned long long)(size_t)xh + (unsigned long long)m0 * K_DIM * 2;
  const unsigned long long wga =
      (unsigned long long)(size_t)wh + (unsigned long long)n0 * K_DIM * 2;

  auto issue_stage = [&](int s) {
    const unsigned bo = (unsigned)(s & 3) * (unsigned)TILE_BYTES;
    const unsigned long long ko = (unsigned long long)s * (BK * 2);
    tdm_load_2d(xlds + bo, xga + ko, K_DIM, 1u << 20, BK, BM, K_DIM);
    tdm_load_2d(wlds + bo, wga + ko, K_DIM, 1u << 20, BK, BN, K_DIM);
  };

  const int nk = K_DIM / BK;              // 64 K-tiles

  if (wid == 0) {                         // TDM ignores EXEC; one wave drives
    issue_stage(0);
    issue_stage(1);
    issue_stage(2);
  }

  for (int kt = 0; kt < nk; ++kt) {
    if (wid == 0) {
      // in-order TENSORcnt: leave the 2 newest stages (4 ops) in flight
      if (kt + 2 < nk)      __builtin_amdgcn_s_wait_tensorcnt(4);
      else if (kt + 1 < nk) __builtin_amdgcn_s_wait_tensorcnt(2);
      else                  __builtin_amdgcn_s_wait_tensorcnt(0);
    }
    __syncthreads();                      // stage kt visible to all waves
    if (wid == 0 && kt + 3 < nk)
      issue_stage(kt + 3);                // buffer freed by last barrier
    const int buf = kt & 3;
    compute_tile(&xsm[buf][0], &wsm[buf][0], wm, wn, half, r, acc);
  }

  store_tile(out, bias, m0, n0, wm, wn, half, r, acc);
}

// ============================================================
// Pass 2 (alternate, compiled-in): async-to-LDS double-buffered GEMM
// ============================================================
__global__ __launch_bounds__(256, 2)
void gemm_f16_wmma_async_kernel(const _Float16* __restrict__ xh,
                                const _Float16* __restrict__ wh,
                                const float* __restrict__ bias,
                                float* __restrict__ out, int M) {
  __shared__ __align__(32) _Float16 xsm[2][BM * BK];
  __shared__ __align__(32) _Float16 wsm[2][BN * BK];

  const int tid  = threadIdx.x;
  const int m0   = blockIdx.y * BM;
  const int n0   = blockIdx.x * BN;
  const int lane = tid & 31;
  const int wid  = tid >> 5;
  const int wm   = wid >> 2;
  const int wn   = wid & 3;
  const int half = lane >> 4;
  const int r    = lane & 15;

  v8f acc[4][2];
  const v8f vzero = {0.f, 0.f, 0.f, 0.f, 0.f, 0.f, 0.f, 0.f};
#pragma unroll
  for (int i = 0; i < 4; ++i)
#pragma unroll
    for (int j = 0; j < 2; ++j)
      acc[i][j] = vzero;

  unsigned xvoff[4], wvoff[4], xldso[4], wldso[4];
  const unsigned xbase = lds_off32(&xsm[0][0]);
  const unsigned wbase = lds_off32(&wsm[0][0]);
#pragma unroll
  for (int i = 0; i < 4; ++i) {
    const int c    = tid + i * 256;
    const int row  = c >> 3;
    const int col8 = (c & 7) << 3;
    xvoff[i] = (unsigned)(((m0 + row) * K_DIM + col8) * 2);
    wvoff[i] = (unsigned)(((n0 + row) * K_DIM + col8) * 2);
    const unsigned t = (unsigned)((row * BK + col8) * 2);
    xldso[i] = xbase + t;
    wldso[i] = wbase + t;
  }

  const int nk = K_DIM / BK;
#pragma unroll
  for (int i = 0; i < 4; ++i) {
    async_load_b128(xldso[i], xvoff[i], xh);
    async_load_b128(wldso[i], wvoff[i], wh);
    xvoff[i] += BK * 2;
    wvoff[i] += BK * 2;
  }
  wait_asynccnt0();
  __syncthreads();

  for (int kt = 0; kt < nk; ++kt) {
    const int buf = kt & 1;
    if (kt + 1 < nk) {
      const unsigned bo = (unsigned)((buf ^ 1) * TILE_BYTES);
#pragma unroll
      for (int i = 0; i < 4; ++i) {
        async_load_b128(xldso[i] + bo, xvoff[i], xh);
        async_load_b128(wldso[i] + bo, wvoff[i], wh);
        xvoff[i] += BK * 2;
        wvoff[i] += BK * 2;
      }
    }
    compute_tile(&xsm[buf][0], &wsm[buf][0], wm, wn, half, r, acc);
    wait_asynccnt0();
    __syncthreads();
  }

  store_tile(out, bias, m0, n0, wm, wn, half, r, acc);
}

// ============================================================
// Fallback: fused dequant-in-LDS kernel (d_ws too small)
// ============================================================
__global__ __launch_bounds__(256, 2)
void int4_linear_fused_kernel(const float* __restrict__ x,
                              const int*   __restrict__ wp,
                              const float* __restrict__ scale,
                              const float* __restrict__ zp,
                              const float* __restrict__ bias,
                              float* __restrict__ out,
                              int M) {
  __shared__ __align__(32) unsigned xbuf[BM * BK / 2];
  __shared__ __align__(32) unsigned wbuf[BN * BK / 2];

  const int tid  = threadIdx.x;
  const int m0   = blockIdx.y * BM;
  const int n0   = blockIdx.x * BN;
  const int lane = tid & 31;
  const int wid  = tid >> 5;
  const int wm   = wid >> 2;
  const int wn   = wid & 3;
  const int half = lane >> 4;
  const int r    = lane & 15;

  v8f acc[4][2];
  const v8f vzero = {0.f, 0.f, 0.f, 0.f, 0.f, 0.f, 0.f, 0.f};
#pragma unroll
  for (int i = 0; i < 4; ++i)
#pragma unroll
    for (int j = 0; j < 2; ++j)
      acc[i][j] = vzero;

  const int wrow  = tid >> 1;
  const int whalf = tid & 1;
  const float* srow  = scale + (size_t)(n0 + wrow) * NG_DIM;
  const float* zrow  = zp    + (size_t)(n0 + wrow) * NG_DIM;
  const int*   wrowp = wp    + (size_t)(n0 + wrow) * PK_DIM;

  const int nk = K_DIM / BK;
  for (int kt = 0; kt < nk; ++kt) {
    const int k0 = kt * BK;
#pragma unroll
    for (int i = 0; i < 8; ++i) {
      const int f   = tid + i * 256;
      const int row = f >> 4;
      const int c4  = (f & 15) << 2;
      const float4 v = *(const float4*)(x + (size_t)(m0 + row) * K_DIM + k0 + c4);
      *(uint2*)&xbuf[(row * BK + c4) >> 1] =
          make_uint2(pack2_f16(v.x, v.y), pack2_f16(v.z, v.w));
    }
    {
      const int   g  = k0 >> 7;
      const float s  = srow[g];
      const float z  = zrow[g];
      const int   j0 = (k0 >> 1) + whalf * 16;
#pragma unroll
      for (int q = 0; q < 4; ++q) {
        const int4 pk = *(const int4*)(wrowp + j0 + q * 4);
        const int pv[4] = {pk.x, pk.y, pk.z, pk.w};
        unsigned o[4];
#pragma unroll
        for (int e = 0; e < 4; ++e) {
          const float hi = (float)((pv[e] >> 4) & 15);
          const float lo = (float)(pv[e] & 15);
          o[e] = pack2_f16((hi - z) * s, (lo - z) * s);
        }
        *(uint4*)&wbuf[(wrow * BK + whalf * 32 + q * 8) >> 1] =
            make_uint4(o[0], o[1], o[2], o[3]);
      }
    }
    __syncthreads();
    compute_tile((const _Float16*)xbuf, (const _Float16*)wbuf,
                 wm, wn, half, r, acc);
    __syncthreads();
  }

  store_tile(out, bias, m0, n0, wm, wn, half, r, acc);
}

extern "C" void kernel_launch(void* const* d_in, const int* in_sizes, int n_in,
                              void* d_out, int out_size, void* d_ws, size_t ws_size,
                              hipStream_t stream) {
  const float* x  = (const float*)d_in[0];
  const int*   wp = (const int*)d_in[1];
  const float* sc = (const float*)d_in[2];
  const float* zp = (const float*)d_in[3];
  const float* bs = (const float*)d_in[4];
  float* out = (float*)d_out;

  const int M = in_sizes[0] / K_DIM;                 // 8192
  const size_t wbytes = (size_t)N_DIM * K_DIM * 2;   // 90.2 MB f16 W
  const size_t xbytes = (size_t)M * K_DIM * 2;       // 67.1 MB f16 x

  dim3 grid(N_DIM / BN, M / BM);                     // 86 x 64
  dim3 block(256);                                   // 8 wave32

  if (ws_size >= wbytes + xbytes) {
    _Float16* w16 = (_Float16*)d_ws;
    _Float16* x16 = (_Float16*)((char*)d_ws + wbytes);
    hipLaunchKernelGGL(dequant_w_kernel, dim3(N_DIM), block, 0, stream,
                       wp, sc, zp, w16);
    hipLaunchKernelGGL(cvt_x_kernel,
                       dim3((unsigned)(((size_t)M * K_DIM) / 2048)), block, 0,
                       stream, x, x16);
    hipLaunchKernelGGL(gemm_f16_wmma_tdm_kernel, grid, block, 0, stream,
                       x16, w16, bs, out, M);
  } else {
    hipLaunchKernelGGL(int4_linear_fused_kernel, grid, block, 0, stream,
                       x, wp, sc, zp, bs, out, M);
  }
}
